// CreateGraphCondensationEdges_23562190586025
// MI455X (gfx1250) — matrix operand order repaired and checked by reading, not compiled, not run on hardware
//
#include <hip/hip_runtime.h>
#include <hip/hip_bf16.h>

// Problem constants (match reference)
#define NN      400000
#define F_IN    64
#define PRE     32
#define KNB     6
#define XN_K    198          // K*(PRE+1)
#define XN_KP   224          // padded to 7 chunks of 32
#define TILES   5            // node tiles per wave per block (edge pass)

typedef __attribute__((ext_vector_type(16))) _Float16 v16h;
typedef __attribute__((ext_vector_type(8)))  _Float16 v8h;
typedef __attribute__((ext_vector_type(8)))  float    v8f;

__device__ __forceinline__ v8f wmma_f16(v16h a, v16h b, v8f c) {
  // D = A(16x32 f16) * B(32x16 f16) + C(16x16 f32)
  return __builtin_amdgcn_wmma_f32_16x16x32_f16(
      /*neg_a=*/false, a, /*neg_b=*/false, b,
      /*c_mod=*/(short)0, c, /*reuse_a=*/false, /*reuse_b=*/false);
}

__device__ __forceinline__ float elu_f(float v) {
  return v > 0.f ? v : (__expf(v) - 1.f);
}

// --- Fragment-layout LDS regions: 512 halves per fragment; each lane's 16
// --- elements contiguous (aligned v16h = 2x ds_load_b128 / ds_store_b128).
//
// A (16x32 f16) ISA layout: lane = m + khalf*16;
//   khalf=0: e 0..7 -> K 0..7,  e 8..15 -> K 16..23
//   khalf=1: e 0..7 -> K 8..15, e 8..15 -> K 24..31
__device__ __forceinline__ int afrag_pos(int m, int kk) {
  const int khalf = (kk >> 3) & 1;
  const int e     = (kk & 7) + ((kk >> 4) << 3);
  return ((m + (khalf << 4)) << 4) + e;
}
// B (32x16 f16): lane = n + ((kk>>3)&1)*16, elem = (kk&7) + (kk>>4)*8
__device__ __forceinline__ int bfrag_pos(int kk, int n) {
  return ((n + (((kk >> 3) & 1) << 4)) << 4) + (kk & 7) + ((kk >> 4) << 3);
}

__device__ __forceinline__ v16h load_frag(const _Float16* base) {
  return *(const v16h*)(base + (threadIdx.x & 31) * 16);
}

// 8 consecutive f32 -> 8 f16 (global_load_b128 x2 + v_cvt_pk_f16_f32 x4)
__device__ __forceinline__ v8h cvt8(const float* p) {
  const v8f f = *(const v8f*)p;
  return __builtin_convertvector(f, v8h);
}
__device__ __forceinline__ v16h cat8(v8h lo, v8h hi) {
  return __builtin_shufflevector(lo, hi, 0, 1, 2, 3, 4, 5, 6, 7,
                                 8, 9, 10, 11, 12, 13, 14, 15);
}

// ---------------------------------------------------------------------------
// Pass 1: x_p = elu(x@Wp+bp) -> f16 workspace (25.6 MB, L2-resident)
//         xs  = selfMLP(x)   -> f32 workspace
// 256 threads = 8 waves, 16 nodes/wave, 128 nodes/block, 3125 blocks.
// A fragments of x built directly from global (no LDS staging).
// ---------------------------------------------------------------------------
__global__ void __launch_bounds__(256) gcn_prepass(
    const float* __restrict__ x,
    const float* __restrict__ Wp,  const float* __restrict__ bp,
    const float* __restrict__ Ws1, const float* __restrict__ bs1,
    const float* __restrict__ Ws2, const float* __restrict__ bs2,
    const float* __restrict__ Ws3, const float* __restrict__ bs3,
    _Float16* __restrict__ xp_out, float* __restrict__ xs_out)
{
  const int tid  = threadIdx.x;
  const int lane = tid & 31;
  const int wave = tid >> 5;
  const int hi   = (lane >> 4) & 1;
  const int nlo  = lane & 15;
  const int node0 = blockIdx.x * 128 + wave * 16;

  __shared__ __align__(32) _Float16 sWpF[4 * 512];     // 4 KB (kc*2+nt)
  __shared__ __align__(32) _Float16 sWs1F[4 * 512];    // 4 KB
  __shared__ __align__(32) _Float16 sWs2F[2 * 512];    // 2 KB
  __shared__ __align__(32) _Float16 sWs3F[512];        // 1 KB
  __shared__ __align__(32) _Float16 sHF[8 * 512];      // 8 KB hidden A-frags
  __shared__ float sBp[32], sBs1[32], sBs2[32];

  // Stage weights into B-frag layout
  for (int i = tid; i < 64 * 32; i += 256) {
    const int r = i >> 5, c = i & 31;
    const int f   = (r >> 5) * 2 + (c >> 4);
    const int pos = bfrag_pos(r & 31, c & 15);
    sWpF [f * 512 + pos] = (_Float16)Wp[i];
    sWs1F[f * 512 + pos] = (_Float16)Ws1[i];
  }
  for (int i = tid; i < 32 * 32; i += 256) {
    const int r = i >> 5, c = i & 31;
    sWs2F[(c >> 4) * 512 + bfrag_pos(r, c & 15)] = (_Float16)Ws2[i];
  }
  for (int i = tid; i < 32 * 16; i += 256) {
    const int r = i >> 4, c = i & 15;
    sWs3F[bfrag_pos(r, c)] = (_Float16)(c == 0 ? Ws3[r] : 0.f);
  }
  if (tid < 32) { sBp[tid] = bp[tid]; sBs1[tid] = bs1[tid]; sBs2[tid] = bs2[tid]; }

  // Build x A-fragments straight from global: lane (m=nlo, khalf=hi) needs
  // contiguous 8-float runs at K = khalf*8 + {0,16,32,48}.
  const float* xrow = x + (size_t)(node0 + nlo) * F_IN + hi * 8;
  const v16h a0 = cat8(cvt8(xrow),      cvt8(xrow + 16));
  const v16h a1 = cat8(cvt8(xrow + 32), cvt8(xrow + 48));
  __syncthreads();

  // ---- x_p = elu(x@Wp + bp): 4 WMMA ----
#pragma unroll
  for (int nt = 0; nt < 2; ++nt) {
    v8f acc = {};
    acc = wmma_f16(a0, load_frag(&sWpF[(0 + nt) * 512]), acc);
    acc = wmma_f16(a1, load_frag(&sWpF[(2 + nt) * 512]), acc);
    const int col = nt * 16 + nlo;
#pragma unroll
    for (int r = 0; r < 8; ++r)
      xp_out[(size_t)(node0 + r + hi * 8) * 32 + col] =
          (_Float16)elu_f(acc[r] + sBp[col]);
  }

  // ---- self MLP layer 1 ----
  _Float16* hF = &sHF[wave * 512];
#pragma unroll
  for (int nt = 0; nt < 2; ++nt) {
    v8f acc = {};
    acc = wmma_f16(a0, load_frag(&sWs1F[(0 + nt) * 512]), acc);
    acc = wmma_f16(a1, load_frag(&sWs1F[(2 + nt) * 512]), acc);
    const int col = nt * 16 + nlo;
#pragma unroll
    for (int r = 0; r < 8; ++r)
      hF[afrag_pos(r + hi * 8, col)] = (_Float16)elu_f(acc[r] + sBs1[col]);
  }
  __syncthreads();

  // ---- self MLP layer 2 ----
  {
    const v16h ah = load_frag(hF);
    v8f b0 = {}, b1 = {};
    b0 = wmma_f16(ah, load_frag(&sWs2F[0]),   b0);
    b1 = wmma_f16(ah, load_frag(&sWs2F[512]), b1);
#pragma unroll
    for (int r = 0; r < 8; ++r) {
      const int m = r + hi * 8;
      hF[afrag_pos(m, nlo)]      = (_Float16)elu_f(b0[r] + sBs2[nlo]);
      hF[afrag_pos(m, nlo + 16)] = (_Float16)elu_f(b1[r] + sBs2[nlo + 16]);
    }
  }
  __syncthreads();

  // ---- self MLP layer 3: [16,32]@[32,1] (col 0 = xs) ----
  {
    const v16h ah = load_frag(hF);
    v8f acc = {};
    acc = wmma_f16(ah, load_frag(sWs3F), acc);
    if (nlo == 0) {
      const float b3 = bs3[0];
#pragma unroll
      for (int r = 0; r < 8; ++r)
        xs_out[node0 + r + hi * 8] = acc[r] + b3;
    }
  }
}

// ---------------------------------------------------------------------------
// Pass 2: gather from f16 x_p (L2), edge MLP 198->32->32->6 via WMMA,
// softmax over [xs, xn] -> out [N,7] f32.
//
// Edge-MLP input columns are PERMUTED (and We1 rows permuted to match, which
// is free at staging time): diff[k][j] -> col k*32+j, distsq[k] -> col 192+k,
// pad -> 198..223. Each neighbor diff then fills one K-chunk fragment with
// two aligned v16h stores (register shuffles only, no per-element addressing).
//
// 128 threads = 4 waves; TILES=5 tiles of 16 nodes/wave; 320 nodes/block,
// 1250 blocks. ~52 KB LDS.
// ---------------------------------------------------------------------------
__global__ void __launch_bounds__(128) gcn_edgepass(
    const float* __restrict__ distsq, const long long* __restrict__ nidx,
    const float* __restrict__ We1, const float* __restrict__ be1,
    const float* __restrict__ We2, const float* __restrict__ be2,
    const float* __restrict__ We3, const float* __restrict__ be3,
    const _Float16* __restrict__ xp, const float* __restrict__ xs,
    float* __restrict__ out)
{
  const int tid  = threadIdx.x;
  const int lane = tid & 31;
  const int wave = tid >> 5;
  const int hi   = (lane >> 4) & 1;
  const int nlo  = lane & 15;

  __shared__ __align__(32) _Float16 sXNF[4 * 7 * 512]; // 28 KB input A-frags
  __shared__ __align__(32) _Float16 sW1F[14 * 512];    // 14 KB (kc*2+nt)
  __shared__ __align__(32) _Float16 sW2F[2 * 512];     // 2 KB
  __shared__ __align__(32) _Float16 sW3F[512];         // 1 KB
  __shared__ __align__(32) _Float16 sHF[4 * 512];      // 4 KB hidden A-frags
  __shared__ float sB1[32], sB2[32], sB3[16];
  __shared__ float sL[4 * 16 * 8];                     // 2 KB edge logits

  // Stage We1 with the row permutation matching the new input column order.
  for (int i = tid; i < XN_KP * 32; i += 128) {
    const int rp = i >> 5, c = i & 31;     // rp = permuted K row
    float w = 0.f;
    if (rp < 192)      w = We1[((rp >> 5) * 33 + (rp & 31)) * 32 + c]; // diff
    else if (rp < XN_K) w = We1[((rp - 192) * 33 + 32) * 32 + c];      // distsq
    sW1F[((rp >> 5) * 2 + (c >> 4)) * 512 + bfrag_pos(rp & 31, c & 15)] =
        (_Float16)w;
  }
  for (int i = tid; i < 32 * 32; i += 128) {
    const int r = i >> 5, c = i & 31;
    sW2F[(c >> 4) * 512 + bfrag_pos(r, c & 15)] = (_Float16)We2[i];
  }
  for (int i = tid; i < 32 * 16; i += 128) {
    const int r = i >> 4, c = i & 15;
    sW3F[bfrag_pos(r, c)] = (_Float16)(c < KNB ? We3[r * KNB + c] : 0.f);
  }
  if (tid < 32) { sB1[tid] = be1[tid]; sB2[tid] = be2[tid]; }
  if (tid < 16) sB3[tid] = (tid < KNB) ? be3[tid] : 0.f;

  const v16h zero16 = {};
  // Zero the kc=6 fragment once (distsq slots kk 0..5 are overwritten per
  // tile; kk 6..31 stay zero). One aligned v16h store per lane covers it.
  _Float16* xnF = &sXNF[wave * 7 * 512];
  *(v16h*)(&xnF[6 * 512 + lane * 16]) = zero16;
  __syncthreads();

#pragma unroll 1
  for (int t = 0; t < TILES; ++t) {
    const int node0 = blockIdx.x * (64 * TILES) + t * 64 + wave * 16;

    // ---- gather + diff: 96 (node,neighbor) pairs/wave, 3 per lane.
    // x_p rows read as v16h (global_load_b128 pairs, L2-resident);
    // packed-f16 diff; two register shuffles -> two aligned v16h LDS stores.
#pragma unroll
    for (int pp = 0; pp < 3; ++pp) {
      const int p = pp * 32 + lane;
      const int i = p / KNB;                      // node row 0..15
      const int k = p - i * KNB;                  // neighbor 0..5
      const int node = node0 + i;
      const long long nb = nidx[(size_t)node * KNB + k];
      const float dq = distsq[(size_t)node * KNB + k];
      const size_t so = (size_t)node * PRE;
      const size_t go = (size_t)(nb >= 0 ? nb : 0) * PRE;
      const v16h s0 = *(const v16h*)(xp + so);
      const v16h s1 = *(const v16h*)(xp + so + 16);
      v16h g0 = *(const v16h*)(xp + go);
      v16h g1 = *(const v16h*)(xp + go + 16);
      if (nb < 0) { g0 = zero16; g1 = zero16; }
      const v16h d0 = s0 - g0;                    // K 0..15 of this chunk
      const v16h d1 = s1 - g1;                    // K 16..31
      // lane row i   (khalf=0): K {0..7, 16..23}
      // lane row i+16(khalf=1): K {8..15, 24..31}
      const v16h va = __builtin_shufflevector(
          d0, d1, 0, 1, 2, 3, 4, 5, 6, 7, 16, 17, 18, 19, 20, 21, 22, 23);
      const v16h vb = __builtin_shufflevector(
          d0, d1, 8, 9, 10, 11, 12, 13, 14, 15, 24, 25, 26, 27, 28, 29, 30, 31);
      *(v16h*)(&xnF[k * 512 + i * 16])        = va;
      *(v16h*)(&xnF[k * 512 + (i + 16) * 16]) = vb;
      xnF[6 * 512 + i * 16 + k] = (_Float16)dq;   // permuted col 192+k
    }
    __syncthreads();

    // ---- edge layer 1: 7 K-chunks x 2 N-tiles = 14 WMMA ----
    v8f acc0 = {}, acc1 = {};
#pragma unroll
    for (int kc = 0; kc < 7; ++kc) {
      const v16h a = load_frag(&xnF[kc * 512]);
      acc0 = wmma_f16(a, load_frag(&sW1F[(kc * 2 + 0) * 512]), acc0);
      acc1 = wmma_f16(a, load_frag(&sW1F[(kc * 2 + 1) * 512]), acc1);
    }
    _Float16* hF = &sHF[wave * 512];
#pragma unroll
    for (int r = 0; r < 8; ++r) {
      const int m = r + hi * 8;
      hF[afrag_pos(m, nlo)]      = (_Float16)elu_f(acc0[r] + sB1[nlo]);
      hF[afrag_pos(m, nlo + 16)] = (_Float16)elu_f(acc1[r] + sB1[nlo + 16]);
    }
    __syncthreads();

    // ---- edge layer 2: 2 WMMA ----
    {
      const v16h ah = load_frag(hF);
      v8f b0 = {}, b1 = {};
      b0 = wmma_f16(ah, load_frag(&sW2F[0]),   b0);
      b1 = wmma_f16(ah, load_frag(&sW2F[512]), b1);
#pragma unroll
      for (int r = 0; r < 8; ++r) {
        const int m = r + hi * 8;
        hF[afrag_pos(m, nlo)]      = (_Float16)elu_f(b0[r] + sB2[nlo]);
        hF[afrag_pos(m, nlo + 16)] = (_Float16)elu_f(b1[r] + sB2[nlo + 16]);
      }
    }
    __syncthreads();

    // ---- edge layer 3: 1 WMMA, cols 0..5 = logits ----
    {
      const v16h ah = load_frag(hF);
      v8f acc = {};
      acc = wmma_f16(ah, load_frag(sW3F), acc);
      float* lW = &sL[wave * 128];
      if (nlo < KNB) {
#pragma unroll
        for (int r = 0; r < 8; ++r)
          lW[(r + hi * 8) * 8 + nlo] = acc[r] + sB3[nlo];
      }
    }
    __syncthreads();

    // ---- softmax over [xs, xn0..5]: lanes 0..15 own one row each ----
    if (lane < 16) {
      const int node = node0 + lane;
      const float* lW = &sL[wave * 128 + lane * 8];
      float v[KNB + 1];
      v[0] = xs[node];
#pragma unroll
      for (int j = 0; j < KNB; ++j) v[1 + j] = lW[j];
      float mx = v[0];
#pragma unroll
      for (int j = 1; j < KNB + 1; ++j) mx = fmaxf(mx, v[j]);
      float e[KNB + 1], s = 0.f;
#pragma unroll
      for (int j = 0; j < KNB + 1; ++j) { e[j] = __expf(v[j] - mx); s += e[j]; }
      const float inv = 1.f / s;
#pragma unroll
      for (int j = 0; j < KNB + 1; ++j)
        out[(size_t)node * (KNB + 1) + j] = e[j] * inv;
    }
    __syncthreads();  // protect xnF/hF/sL reuse in next tile iteration
  }
}

extern "C" void kernel_launch(void* const* d_in, const int* in_sizes, int n_in,
                              void* d_out, int out_size, void* d_ws, size_t ws_size,
                              hipStream_t stream) {
  (void)in_sizes; (void)n_in; (void)out_size; (void)ws_size;
  const float*     x      = (const float*)d_in[0];
  const float*     distsq = (const float*)d_in[1];
  const long long* nidx   = (const long long*)d_in[2];   // int64 in reference
  const float*     Wp  = (const float*)d_in[3];
  const float*     bp  = (const float*)d_in[4];
  const float*     We1 = (const float*)d_in[5];
  const float*     be1 = (const float*)d_in[6];
  const float*     We2 = (const float*)d_in[7];
  const float*     be2 = (const float*)d_in[8];
  const float*     We3 = (const float*)d_in[9];
  const float*     be3 = (const float*)d_in[10];
  const float*     Ws1 = (const float*)d_in[11];
  const float*     bs1 = (const float*)d_in[12];
  const float*     Ws2 = (const float*)d_in[13];
  const float*     bs2 = (const float*)d_in[14];
  const float*     Ws3 = (const float*)d_in[15];
  const float*     bs3 = (const float*)d_in[16];

  // Workspace: x_p f16 (25.6 MB, fits in 192 MB L2 for the gather) + xs f32
  _Float16* ws_xp = (_Float16*)d_ws;
  float*    ws_xs = (float*)((char*)d_ws + (size_t)NN * PRE * sizeof(_Float16));

  gcn_prepass<<<NN / 128, 256, 0, stream>>>(x, Wp, bp, Ws1, bs1, Ws2, bs2,
                                            Ws3, bs3, ws_xp, ws_xs);
  gcn_edgepass<<<NN / (64 * TILES), 128, 0, stream>>>(
      distsq, nidx, We1, be1, We2, be2, We3, be3, ws_xp, ws_xs, (float*)d_out);
}
